// MoE_cond_39324720562990
// MI455X (gfx1250) — matrix operand order repaired and checked by reading, not compiled, not run on hardware
//
#include <hip/hip_runtime.h>
#include <cmath>

// Problem constants (match reference)
constexpr int TB = 16384; // B tokens
constexpr int TD = 1024;  // D input dim
constexpr int TH = 1024;  // H output dim
constexpr int TE = 16;    // experts
constexpr int TC = 1024;  // cond dim
constexpr int TK = 3;     // top-k

typedef __attribute__((ext_vector_type(2)))  float  v2f;
typedef __attribute__((ext_vector_type(8)))  float  v8f;
typedef __attribute__((ext_vector_type(16))) __bf16 v16bf;
typedef __attribute__((ext_vector_type(8)))  __bf16 v8bf;

#if defined(__has_builtin)
#if __has_builtin(__builtin_amdgcn_wmma_f32_16x16x4_f32)
#define GATE_WMMA 1
#endif
#endif
#ifndef GATE_WMMA
#define GATE_WMMA 0
#endif

// ---------------------------------------------------------------------------
// fp32 -> bf16 conversion (RNE), 8 elements/thread, bandwidth-trivial
// ---------------------------------------------------------------------------
__global__ __launch_bounds__(256) void cvt_f32_bf16(
    const float* __restrict__ in, __bf16* __restrict__ out, int n) {
  int i = (blockIdx.x * 256 + threadIdx.x) * 8;
  if (i >= n) return;
  const float4* p = reinterpret_cast<const float4*>(in + i);
  float4 a = p[0], b = p[1];
  v8bf o;
  o[0] = (__bf16)a.x; o[1] = (__bf16)a.y; o[2] = (__bf16)a.z; o[3] = (__bf16)a.w;
  o[4] = (__bf16)b.x; o[5] = (__bf16)b.y; o[6] = (__bf16)b.z; o[7] = (__bf16)b.w;
  *reinterpret_cast<v8bf*>(out + i) = o;
}

// ---------------------------------------------------------------------------
// bf16 WMMA inner loop: one wave computes a 32(M) x 64(N) f32 tile (K = TD).
// Two A fragments (rows m0..m0+15 and m0+16..m0+31) share every B fragment,
// doubling FLOP per byte of operand traffic vs a 16x64 tile (21.3 FLOP/B).
// Register layouts used here (contiguous k order, same for A and B):
//   A 16x32 bf16: lanes 0-15 own row M=lane,    K in {k..k+15}
//                 lanes 16-31 own row M=lane-16, K in {k+16..k+31}
//   B 32x16 bf16: lane owns column n = lane&15, 16 contiguous K values
// ---------------------------------------------------------------------------
__device__ __forceinline__ void load_a_frag(const __bf16* __restrict__ p, v16bf& a) {
  v8bf c0 = *reinterpret_cast<const v8bf*>(p);
  v8bf c1 = *reinterpret_cast<const v8bf*>(p + 8);
#pragma unroll
  for (int i = 0; i < 8; ++i) { a[i] = c0[i]; a[8 + i] = c1[i]; }
}

__device__ __forceinline__ void gemm_bf16_32x64(
    const __bf16* __restrict__ arow0,  // A row base, lane's row in M-tile 0
    const __bf16* __restrict__ arow1,  // A row base, lane's row in M-tile 1
    const __bf16* __restrict__ bblk,   // weight block base: row n0, row-major [*, TD]
    int lm, int lh, v8f acc[2][4]) {
#pragma unroll 2
  for (int k = 0; k < TD; k += 32) {
    v16bf a0, a1;
    load_a_frag(arow0 + k + 16 * lh, a0);
    load_a_frag(arow1 + k + 16 * lh, a1);
    __builtin_prefetch(arow0 + k + 512, 0, 1);  // global_prefetch_b8, speculative
    __builtin_prefetch(arow1 + k + 512, 0, 1);
#pragma unroll
    for (int t = 0; t < 4; ++t) {
      const __bf16* brow = bblk + (size_t)(t * 16 + lm) * TD + k + 16 * lh;
      v16bf b = *reinterpret_cast<const v16bf*>(brow);
      acc[0][t] = __builtin_amdgcn_wmma_f32_16x16x32_bf16(
          false, a0, false, b, (short)0, acc[0][t], false, false);
      acc[1][t] = __builtin_amdgcn_wmma_f32_16x16x32_bf16(
          false, a1, false, b, (short)0, acc[1][t], false, false);
    }
  }
}

// ---------------------------------------------------------------------------
// Gate: h = GELU(cond @ Wg1^T + bg1); logits = h @ Wg2^T + bg2;
// softmax over E=16; top-3; compact routes per expert.
// 16 tokens per block, 8 waves. GEMM1 in fp32 WMMA (16x16x4) to keep the
// discrete top-k selection numerically faithful. GEMM2 fused: each wave's
// 16x16 h tile bounces through per-wave LDS (C-layout -> A-layout transpose,
// DS ops are in-order within a wave) into more f32 WMMAs.
// ---------------------------------------------------------------------------
__global__ __launch_bounds__(256) void gate_kernel(
    const float* __restrict__ cond, const float* __restrict__ Wg1,
    const float* __restrict__ bg1, const float* __restrict__ Wg2,
    const float* __restrict__ bg2, int* __restrict__ cnt,
    int* __restrict__ rtok, float* __restrict__ rprob) {
  __shared__ float logits[16][TE];
  const int m0 = blockIdx.x * 16;

  // init logits with bg2 (256 threads cover 16x16)
  logits[threadIdx.x >> 4][threadIdx.x & 15] = bg2[threadIdx.x & 15];
  __syncthreads();

#if GATE_WMMA
  __shared__ float hscr[8][16][17];  // per-wave 16x16 transpose scratch (+pad)
  const int wave = threadIdx.x >> 5, lane = threadIdx.x & 31;
  const int lm = lane & 15, lh = lane >> 4;
  v8f lacc = {};  // partial logits (16 tokens x 16 experts)

  for (int t = 0; t < 8; ++t) {
    const int j0 = (wave * 8 + t) * 16;  // 64 N-tiles / 8 waves
    v8f acc = {};
    // fp32 WMMA layouts (16x16x4): A lane owns M=lane&15, K=2*lh+{0,1};
    // B lane owns N=lane&15, K=2*lh+{0,1}
    const float* ar = cond + (size_t)(m0 + lm) * TC + 2 * lh;
    const float* br = Wg1 + (size_t)(j0 + lm) * TC + 2 * lh;
#pragma unroll 4
    for (int k = 0; k < TC; k += 4) {
      v2f av = *reinterpret_cast<const v2f*>(ar + k);
      v2f bv = *reinterpret_cast<const v2f*>(br + k);
      acc = __builtin_amdgcn_wmma_f32_16x16x4_f32(
          false, av, false, bv, (short)0, acc, false, false);
    }
    // bias + exact-erf GELU; C layout: row = r + 8*lh, col = lm
    const float bias = bg1[j0 + lm];
#pragma unroll
    for (int r = 0; r < 8; ++r) {
      float v = acc[r] + bias;
      v = 0.5f * v * (1.0f + erff(v * 0.70710678118654752f));
      hscr[wave][r + 8 * lh][lm] = v;  // stage for transpose
    }
    // GEMM2 partial over this 16-wide j slice: A(m,k)=h[m][j0+k], B(k,e)=Wg2[e][j0+k]
#pragma unroll
    for (int kk = 0; kk < 16; kk += 4) {
      v2f a2, b2;
      a2[0] = hscr[wave][lm][kk + 2 * lh + 0];
      a2[1] = hscr[wave][lm][kk + 2 * lh + 1];
      b2[0] = Wg2[(size_t)lm * TC + j0 + kk + 2 * lh + 0];
      b2[1] = Wg2[(size_t)lm * TC + j0 + kk + 2 * lh + 1];
      lacc = __builtin_amdgcn_wmma_f32_16x16x4_f32(
          false, a2, false, b2, (short)0, lacc, false, false);
    }
  }
  // reduce partial logits across waves (ds_add_f32)
#pragma unroll
  for (int r = 0; r < 8; ++r)
    __hip_atomic_fetch_add(&logits[r + 8 * lh][lm], lacc[r],
                           __ATOMIC_RELAXED, __HIP_MEMORY_SCOPE_WORKGROUP);
#else
  // Scalar fallback (host pass / toolchains without the f32 WMMA builtin)
  for (int m = 0; m < 16; ++m) {
    float part[TE] = {};
    for (int j = threadIdx.x; j < TC; j += 256) {
      float hv = 0.f;
      for (int k = 0; k < TC; ++k) hv += cond[(size_t)(m0 + m) * TC + k] * Wg1[(size_t)j * TC + k];
      hv += bg1[j];
      hv = 0.5f * hv * (1.0f + erff(hv * 0.70710678118654752f));
      for (int e = 0; e < TE; ++e) part[e] += hv * Wg2[(size_t)e * TC + j];
    }
    for (int e = 0; e < TE; ++e)
      __hip_atomic_fetch_add(&logits[m][e], part[e],
                             __ATOMIC_RELAXED, __HIP_MEMORY_SCOPE_WORKGROUP);
  }
#endif
  __syncthreads();

  // softmax (full E) + top-3 (ties resolved to lowest index, like lax.top_k)
  if (threadIdx.x < 16) {
    const int m = threadIdx.x;
    float l[TE];
    float mx = -3.0e38f;
#pragma unroll
    for (int e = 0; e < TE; ++e) { l[e] = logits[m][e]; mx = fmaxf(mx, l[e]); }
    float s = 0.f;
#pragma unroll
    for (int e = 0; e < TE; ++e) { l[e] = expf(l[e] - mx); s += l[e]; }
    const float inv = 1.0f / s;
    int ex0 = -1, ex1 = -1;
    for (int kk = 0; kk < TK; ++kk) {
      int best = 0; float bv = -1.0f;
      for (int e = 0; e < TE; ++e) {
        if (e == ex0 || e == ex1) continue;
        if (l[e] > bv) { bv = l[e]; best = e; }
      }
      const int pos = atomicAdd(&cnt[best], 1);
      rtok[best * TB + pos] = m0 + m;
      rprob[best * TB + pos] = bv * inv;
      if (kk == 0) ex0 = best; else ex1 = best;
    }
  }
}

// ---------------------------------------------------------------------------
// Generalist: out = x @ Wgen^T + bgen (initializes the output buffer)
// Block tile 64(M) x 256(N): 8 waves in 2x4, each wave 32x64.
// ---------------------------------------------------------------------------
constexpr int GM = 64, GN = 256;

__global__ __launch_bounds__(256) void gen_kernel(
    const __bf16* __restrict__ xb, const __bf16* __restrict__ Wgb,
    const float* __restrict__ bgen, float* __restrict__ out) {
  const int wave = threadIdx.x >> 5, lane = threadIdx.x & 31;
  const int lm = lane & 15, lh = lane >> 4;
  const int m0 = blockIdx.y * GM + (wave >> 2) * 32;
  const int n0 = blockIdx.x * GN + (wave & 3) * 64;
  v8f acc[2][4] = {};
  gemm_bf16_32x64(xb + (size_t)(m0 + lm) * TD,
                  xb + (size_t)(m0 + 16 + lm) * TD,
                  Wgb + (size_t)n0 * TD, lm, lh, acc);
#pragma unroll
  for (int t = 0; t < 4; ++t) {
    const int col = n0 + t * 16 + lm;
    const float bias = bgen[col];
#pragma unroll
    for (int s = 0; s < 2; ++s)
#pragma unroll
      for (int r = 0; r < 8; ++r)
        out[(size_t)(m0 + s * 16 + r + 8 * lh) * TH + col] = acc[s][t][r] + bias;
  }
}

// ---------------------------------------------------------------------------
// Grouped expert GEMM over the routed token lists.
// grid = (H/GN, B/GM, E); blocks beyond this expert's count exit immediately,
// so only the routed ~K/E fraction of work executes (103 GFLOP total, 5.3x
// less than the dense all-expert reference einsum).
// Epilogue: out[tok] += p * (acc + be[e])  via global_atomic_add_f32.
// ---------------------------------------------------------------------------
__global__ __launch_bounds__(256) void expert_kernel(
    const __bf16* __restrict__ xb, const __bf16* __restrict__ Web,
    const float* __restrict__ be, const int* __restrict__ cnt,
    const int* __restrict__ rtok, const float* __restrict__ rprob,
    float* __restrict__ out) {
  const int e = blockIdx.z;
  const int ne = cnt[e];
  const int mbase = blockIdx.y * GM;
  if (mbase >= ne) return;  // uniform early-exit (EXEC stays all-ones for WMMA)
  const int wave = threadIdx.x >> 5, lane = threadIdx.x & 31;
  const int lm = lane & 15, lh = lane >> 4;
  const int m0 = mbase + (wave >> 2) * 32;
  const int n0 = blockIdx.x * GN + (wave & 3) * 64;

  // gather this lane's two A rows; clamp out-of-range list slots (masked below)
  int r0 = m0 + lm;      if (r0 > ne - 1) r0 = ne - 1;
  int r1 = m0 + 16 + lm; if (r1 > ne - 1) r1 = ne - 1;
  const int tok0 = rtok[e * TB + r0];
  const int tok1 = rtok[e * TB + r1];

  v8f acc[2][4] = {};
  gemm_bf16_32x64(xb + (size_t)tok0 * TD, xb + (size_t)tok1 * TD,
                  Web + ((size_t)e * TH + n0) * TD, lm, lh, acc);
#pragma unroll
  for (int s = 0; s < 2; ++s) {
#pragma unroll
    for (int r = 0; r < 8; ++r) {
      const int li = m0 + s * 16 + r + 8 * lh;
      if (li < ne) {
        const int t2 = rtok[e * TB + li];
        const float p = rprob[e * TB + li];
#pragma unroll
        for (int t = 0; t < 4; ++t) {
          const int col = n0 + t * 16 + lm;
          __hip_atomic_fetch_add(out + (size_t)t2 * TH + col,
                                 p * (acc[s][t][r] + be[(size_t)e * TH + col]),
                                 __ATOMIC_RELAXED, __HIP_MEMORY_SCOPE_AGENT);
        }
      }
    }
  }
}

// ---------------------------------------------------------------------------
extern "C" void kernel_launch(void* const* d_in, const int* in_sizes, int n_in,
                              void* d_out, int out_size, void* d_ws, size_t ws_size,
                              hipStream_t stream) {
  (void)in_sizes; (void)n_in; (void)out_size; (void)ws_size;
  const float* x    = (const float*)d_in[0];
  const float* cond = (const float*)d_in[1];
  const float* We   = (const float*)d_in[2];
  const float* be   = (const float*)d_in[3];
  const float* Wg1  = (const float*)d_in[4];
  const float* bg1  = (const float*)d_in[5];
  const float* Wg2  = (const float*)d_in[6];
  const float* bg2  = (const float*)d_in[7];
  const float* Wgen = (const float*)d_in[8];
  const float* bgen = (const float*)d_in[9];
  float* out = (float*)d_out;

  // workspace layout (~71.3 MB)
  char* ws = (char*)d_ws;
  size_t off = 0;
  __bf16* xb   = (__bf16*)(ws + off); off += (size_t)TB * TD * 2;
  __bf16* Web  = (__bf16*)(ws + off); off += (size_t)TE * TH * TD * 2;
  __bf16* Wgb  = (__bf16*)(ws + off); off += (size_t)TH * TD * 2;
  int*    cnt  = (int*)(ws + off);    off += 256;
  int*    rtok = (int*)(ws + off);    off += (size_t)TE * TB * 4;
  float*  rprob= (float*)(ws + off);  off += (size_t)TE * TB * 4;

  hipMemsetAsync(cnt, 0, TE * sizeof(int), stream);

  const int nx = TB * TD, nwe = TE * TH * TD, nwg = TH * TD;
  cvt_f32_bf16<<<nx  / 2048, 256, 0, stream>>>(x, xb, nx);
  cvt_f32_bf16<<<nwe / 2048, 256, 0, stream>>>(We, Web, nwe);
  cvt_f32_bf16<<<nwg / 2048, 256, 0, stream>>>(Wgen, Wgb, nwg);

  gate_kernel<<<TB / 16, 256, 0, stream>>>(cond, Wg1, bg1, Wg2, bg2, cnt, rtok, rprob);
  gen_kernel<<<dim3(TH / GN, TB / GM), 256, 0, stream>>>(xb, Wgb, bgen, out);
  expert_kernel<<<dim3(TH / GN, TB / GM, TE), 256, 0, stream>>>(
      xb, Web, be, cnt, rtok, rprob, out);
}
